// SlidingATTN_31301721653320
// MI455X (gfx1250) — matrix operand order repaired and checked by reading, not compiled, not run on hardware
//
#include <hip/hip_runtime.h>
#include <hip/hip_bf16.h>

typedef __attribute__((ext_vector_type(16))) _Float16 v16h;
typedef __attribute__((ext_vector_type(8)))  _Float16 v8h;
typedef __attribute__((ext_vector_type(8)))  float    v8f;
typedef __attribute__((ext_vector_type(4)))  unsigned int u32x4;
typedef __attribute__((ext_vector_type(8)))  int      i32x8;
typedef __attribute__((ext_vector_type(4)))  int      i32x4;

#define Bb   16
#define Ll   384
#define DIMc 512
#define Hh   8
#define Ww   31
#define DHc  64
#define Mrows (Bb * Ll)   // 6144

#if __has_builtin(__builtin_amdgcn_tensor_load_to_lds) && \
    __has_builtin(__builtin_amdgcn_s_wait_tensorcnt)
#define USE_TDM 1
#else
#define USE_TDM 0
#endif

// ---------------------------------------------------------------------------
// fp32 -> f16 conversion (one pass per tensor; amortized across 384 GEMM blocks)
// ---------------------------------------------------------------------------
__global__ void k_cvt_f16(const float* __restrict__ src,
                          _Float16* __restrict__ dst, int n) {
    for (int i = blockIdx.x * blockDim.x + threadIdx.x; i < n;
         i += gridDim.x * blockDim.x)
        dst[i] = (_Float16)src[i];
}

// ---------------------------------------------------------------------------
// K1: attn_exp[b,l,h] = exp(x[b,l,:]·w_attn[:,h] + b_attn[h]) * (!mask)
// ---------------------------------------------------------------------------
__global__ void k_attn_exp(const float* __restrict__ x,
                           const unsigned char* __restrict__ mask,
                           const float* __restrict__ w_attn,
                           const float* __restrict__ b_attn,
                           float* __restrict__ attn_exp) {
    int bl   = blockIdx.x;
    int h    = threadIdx.x >> 5;
    int lane = threadIdx.x & 31;
    const float* xr = x + (size_t)bl * DIMc;
    float p = 0.f;
    for (int i = lane; i < DIMc; i += 32)
        p += xr[i] * w_attn[i * Hh + h];
    for (int off = 16; off > 0; off >>= 1)
        p += __shfl_down(p, off, 32);
    if (lane == 0) {
        float val = mask[bl] ? 0.f : __expf(p + b_attn[h]);
        attn_exp[bl * Hh + h] = val;
    }
}

// ---------------------------------------------------------------------------
// K2: denom[b,h] = sum_l attn_exp[b,l,h]
// ---------------------------------------------------------------------------
__global__ void k_denom(const float* __restrict__ attn_exp,
                        float* __restrict__ denom) {
    int b = blockIdx.x >> 3, h = blockIdx.x & 7;
    __shared__ float red[256];
    float s = 0.f;
    for (int l = threadIdx.x; l < Ll; l += 256)
        s += attn_exp[(b * Ll + l) * Hh + h];
    red[threadIdx.x] = s;
    __syncthreads();
    for (int st = 128; st > 0; st >>= 1) {
        if (threadIdx.x < st) red[threadIdx.x] += red[threadIdx.x + st];
        __syncthreads();
    }
    if (threadIdx.x == 0) denom[blockIdx.x] = red[0];
}

// ---------------------------------------------------------------------------
// TDM helper: async-copy a contiguous [rows x row_elems] f16 tile to LDS.
// D# packing per CDNA5 ISA:
//   group0: count=1, lds_addr, 57-bit global_addr, type=2
//   group1: data_size=1 (2B), tensor_dim0=row_elems, tensor_dim1=rows,
//           tile_dim0=row_elems, tile_dim1=rows, dim0_stride=row_elems
// ---------------------------------------------------------------------------
#if USE_TDM
__device__ __forceinline__ void tdm_load_tile_f16(const _Float16* gsrc,
                                                  unsigned lds_addr,
                                                  unsigned rows,
                                                  unsigned row_elems) {
    unsigned long long ga = (unsigned long long)(uintptr_t)gsrc;
    u32x4 g0;
    g0[0] = 1u;                                        // count=1 (valid D#)
    g0[1] = lds_addr;                                  // LDS byte address
    g0[2] = (unsigned)(ga & 0xffffffffu);              // global_addr[31:0]
    g0[3] = (unsigned)((ga >> 32) & 0x01ffffffu)       // global_addr[56:32]
            | (2u << 30);                              // type = 2 ("image")
    i32x8 g1;
    g1[0] = (int)(1u << 16);                           // data_size = 2 bytes
    g1[1] = (int)(row_elems << 16);                    // tensor_dim0[15:0]
    g1[2] = (int)(rows << 16);                         // tensor_dim1[15:0]
    g1[3] = (int)(row_elems << 16);                    // tile_dim0
    g1[4] = (int)rows;                                 // tile_dim1 (tile_dim2=0)
    g1[5] = (int)row_elems;                            // tensor_dim0_stride[31:0]
    g1[6] = 0;
    g1[7] = 0;
    i32x4 z4 = {0, 0, 0, 0};
#if __clang_major__ >= 23
    i32x8 z8 = {0, 0, 0, 0, 0, 0, 0, 0};
    __builtin_amdgcn_tensor_load_to_lds(g0, g1, z4, z4, z8, 0);
#else
    __builtin_amdgcn_tensor_load_to_lds(g0, g1, z4, z4, 0);
#endif
}
#endif

// ---------------------------------------------------------------------------
// Fused GEMM + bias + LayerNorm + ELU (+ optional residual).
// C[M,512] = Ah[M,512](f16) @ Wh[512,512](f16); block = 16 rows x 512 cols.
// A tile (16x512 f16 = 16KB contiguous) staged to LDS via the Tensor Data
// Mover (TENSORcnt + s_wait_tensorcnt), fallback: manual vector copy.
// ---------------------------------------------------------------------------
__global__ void gemm_ln_elu(const _Float16* __restrict__ Ah,
                            const _Float16* __restrict__ Wh,
                            const float* __restrict__ bias,
                            const float* __restrict__ gamma,
                            const float* __restrict__ beta,
                            const float* __restrict__ resid,  // may be null
                            float* __restrict__ out) {
    __shared__ _Float16 As[16 * DIMc];      // f16 A tile (contiguous for TDM)
    __shared__ float    Cs[16][DIMc + 4];   // f32 accum tile
    __shared__ float    red[2][16][16];

    const int row0 = blockIdx.x * 16;
    const int tid  = threadIdx.x;

#if USE_TDM
    if (tid == 0) {
        tdm_load_tile_f16(Ah + (size_t)row0 * DIMc,
                          (unsigned)(uintptr_t)(void*)As, 16, DIMc);
        __builtin_amdgcn_s_wait_tensorcnt(0);
    }
#else
    {   // manual staging: 16KB as 64-bit chunks
        const unsigned long long* gs =
            (const unsigned long long*)(Ah + (size_t)row0 * DIMc);
        unsigned long long* ls = (unsigned long long*)As;
        for (int idx = tid; idx < 16 * DIMc / 4; idx += 256)
            ls[idx] = gs[idx];
    }
#endif
    __syncthreads();

    const int wave = tid >> 5;
    const int lane = tid & 31;
    const int m    = lane & 15;
    const int hs   = lane >> 4;   // half-wave select

    // Per ISA A-layout, a lane's fragment is two contiguous 8-half chunks:
    //   k = k0 + hs*8 + [0..7]  and  k0 + 16 + hs*8 + [0..7]
    const _Float16* arow = As + m * DIMc + (hs << 3);

    for (int t = 0; t < 4; ++t) {
        const int n0 = (wave * 4 + t) * 16;
        v8f c = {};
        for (int k0 = 0; k0 < DIMc; k0 += 32) {
            v8h alo = *(const v8h*)(arow + k0);
            v8h ahi = *(const v8h*)(arow + k0 + 16);
            v16h a = __builtin_shufflevector(alo, ahi,
                         0, 1, 2, 3, 4, 5, 6, 7, 8, 9, 10, 11, 12, 13, 14, 15);
            // B fragment: col n = n0 + (lane&15), k = k0 + hs*8 + [0..7] and k0 + 16 + hs*8 + [0..7]
            const _Float16* wr = Wh + (size_t)(k0 + (hs << 3)) * DIMc + n0 + m;
            v16h bfrag;
#pragma unroll
            for (int i = 0; i < 8; ++i) {
                bfrag[i]     = wr[(size_t)i * DIMc];
                bfrag[8 + i] = wr[(size_t)(16 + i) * DIMc];
            }
            c = __builtin_amdgcn_wmma_f32_16x16x32_f16(
                    false, a, false, bfrag, (short)0, c, false, false);
        }
        // C layout: row = r + 8*hs, col = n0 + (lane&15)
#pragma unroll
        for (int r = 0; r < 8; ++r) {
            int row = r + (hs << 3);
            int col = n0 + (lane & 15);
            Cs[row][col] = c[r] + bias[col];
        }
    }
    __syncthreads();

    // LayerNorm + ELU (+ residual) over each 512-wide row
    const int rr = tid >> 4;   // row 0..15
    const int cg = tid & 15;   // 16 column groups
    float s = 0.f, ss = 0.f;
    for (int c = cg; c < DIMc; c += 16) {
        float v0 = Cs[rr][c];
        s += v0; ss += v0 * v0;
    }
    red[0][rr][cg] = s; red[1][rr][cg] = ss;
    __syncthreads();
    if (cg == 0) {
        float S = 0.f, SS = 0.f;
        for (int i = 0; i < 16; ++i) { S += red[0][rr][i]; SS += red[1][rr][i]; }
        float mean = S * (1.f / DIMc);
        float var  = SS * (1.f / DIMc) - mean * mean;
        red[0][rr][0] = mean;
        red[1][rr][0] = rsqrtf(var + 1e-5f);
    }
    __syncthreads();
    const float mean = red[0][rr][0], rstd = red[1][rr][0];
    const float* res = resid ? resid + (size_t)(row0 + rr) * DIMc : nullptr;
    float* orow = out + (size_t)(row0 + rr) * DIMc;
    for (int c = cg; c < DIMc; c += 16) {
        float y = (Cs[rr][c] - mean) * rstd * gamma[c] + beta[c];
        y = y > 0.f ? y : (__expf(y) - 1.f);          // ELU
        if (res) y += res[c];
        orow[c] = y;
    }
}

// ---------------------------------------------------------------------------
// K4: g_pool[b,h,d] = sum_l (attn_exp/denom')[b,l,h] * v[b,l,h*64+d]
// ---------------------------------------------------------------------------
__global__ void k_gpool(const float* __restrict__ attn_exp,
                        const float* __restrict__ denom,
                        const float* __restrict__ v,
                        float* __restrict__ gpool) {
    int b = blockIdx.x >> 3, h = blockIdx.x & 7;
    int d = threadIdx.x & 63, chunk = threadIdx.x >> 6;
    __shared__ float red[256];
    float s = 0.f;
    for (int l = chunk; l < Ll; l += 4)
        s += attn_exp[(b * Ll + l) * Hh + h] *
             v[(size_t)(b * Ll + l) * DIMc + h * DHc + d];
    red[threadIdx.x] = s;
    __syncthreads();
    if (chunk == 0) {
        float tot = red[d] + red[64 + d] + red[128 + d] + red[192 + d];
        float invden = 1.f / (denom[blockIdx.x] + 1e-5f);
        gpool[b * DIMc + h * DHc + d] = tot * invden;
    }
}

// ---------------------------------------------------------------------------
// K5: per-batch positional MLP -> pos[b, (W+1)*H] gates
// ---------------------------------------------------------------------------
__global__ void k_pos(const float* __restrict__ gpool,
                      const float* __restrict__ w_pos1,
                      const float* __restrict__ b_pos1,
                      const float* __restrict__ g_pos,
                      const float* __restrict__ bb_pos,
                      const float* __restrict__ w_pos2,
                      const float* __restrict__ b_pos2,
                      float* __restrict__ posb) {
    int b = blockIdx.x;
    __shared__ float q[8], tt[8], stats[2];
    int h = threadIdx.x >> 5, lane = threadIdx.x & 31;
    const float* gr = gpool + b * DIMc;
    float p = 0.f;
    for (int i = lane; i < DIMc; i += 32) p += gr[i] * w_pos1[i * Hh + h];
    for (int off = 16; off > 0; off >>= 1) p += __shfl_down(p, off, 32);
    if (lane == 0) q[h] = p + b_pos1[h];
    __syncthreads();
    if (threadIdx.x == 0) {
        float m = 0.f;
        for (int i = 0; i < 8; ++i) m += q[i];
        m *= 0.125f;
        float v2 = 0.f;
        for (int i = 0; i < 8; ++i) { float d = q[i] - m; v2 += d * d; }
        v2 *= 0.125f;
        stats[0] = m; stats[1] = rsqrtf(v2 + 1e-5f);
    }
    __syncthreads();
    if (threadIdx.x < 8) {
        float y = (q[threadIdx.x] - stats[0]) * stats[1] * g_pos[threadIdx.x]
                  + bb_pos[threadIdx.x];
        tt[threadIdx.x] = y > 0.f ? y : (__expf(y) - 1.f);
    }
    __syncthreads();
    int j = threadIdx.x;  // (W+1)*H == 256
    float pp = b_pos2[j];
    for (int hh = 0; hh < 8; ++hh) pp += tt[hh] * w_pos2[hh * 256 + j];
    posb[b * 256 + j] = __expf(pp);
}

// ---------------------------------------------------------------------------
// K6: sliding-window aggregation (faithful to reference's slot-0 quirk).
// Writes f16 (direct A-operand of the second WMMA GEMM).
// ---------------------------------------------------------------------------
__global__ void k_window(const float* __restrict__ attn_exp,
                         const float* __restrict__ posb,
                         const float* __restrict__ v,
                         const float* __restrict__ gpool,
                         _Float16* __restrict__ opre_h) {
    int b = blockIdx.x / Ll, l = blockIdx.x % Ll;
    __shared__ float wt[8][32];
    __shared__ float nrm[8];
    int h = threadIdx.x >> 5, s = threadIdx.x & 31;

    float pv = posb[b * 256 + s * Hh + h];
    float val;
    if (s == 0) {
        val = pv;
    } else {
        int j = l + (s - 1) - (Ww / 2);
        val = (j >= 0 && j < Ll) ? attn_exp[(b * Ll + j) * Hh + h] * pv : 0.f;
    }
    wt[h][s] = val;
    __syncthreads();
    if (s == 0) {
        float sum = 1e-5f;
        for (int i = 0; i < 32; ++i) sum += wt[h][i];
        nrm[h] = 1.f / sum;
    }
    __syncthreads();
    wt[h][s] = val * nrm[h];
    __syncthreads();

    int dg = threadIdx.x * 2;           // 2 output dims per thread
    int hh = dg >> 6;
    float acc0 = 0.f, acc1 = 0.f;
    for (int w = 0; w < Ww; ++w) {
        int j = l + w - (Ww / 2);
        if (j < 0 || j >= Ll) continue;
        float wv = wt[hh][w];
        const float* vr = v + (size_t)(b * Ll + j) * DIMc + dg;
        acc0 += wv * vr[0];
        acc1 += wv * vr[1];
    }
    float g0 = wt[hh][0];
    acc0 += g0 * gpool[b * DIMc + dg];
    acc1 += g0 * gpool[b * DIMc + dg + 1];
    _Float16* orow = opre_h + (size_t)(b * Ll + l) * DIMc + dg;
    orow[0] = (_Float16)acc0;
    orow[1] = (_Float16)acc1;
}

// ---------------------------------------------------------------------------
extern "C" void kernel_launch(void* const* d_in, const int* in_sizes, int n_in,
                              void* d_out, int out_size, void* d_ws, size_t ws_size,
                              hipStream_t stream) {
    const float* x       = (const float*)d_in[0];
    const unsigned char* mask = (const unsigned char*)d_in[1];
    const float* w_attn  = (const float*)d_in[2];
    const float* b_attn  = (const float*)d_in[3];
    const float* w_v     = (const float*)d_in[4];
    const float* b_v     = (const float*)d_in[5];
    const float* g_v     = (const float*)d_in[6];
    const float* bb_v    = (const float*)d_in[7];
    const float* w_pos1  = (const float*)d_in[8];
    const float* b_pos1  = (const float*)d_in[9];
    const float* g_pos   = (const float*)d_in[10];
    const float* bb_pos  = (const float*)d_in[11];
    const float* w_pos2  = (const float*)d_in[12];
    const float* b_pos2  = (const float*)d_in[13];
    const float* w_out   = (const float*)d_in[14];
    const float* b_out   = (const float*)d_in[15];
    const float* g_out   = (const float*)d_in[16];
    const float* bb_out  = (const float*)d_in[17];
    float* out = (float*)d_out;

    // fp32 workspace
    float* ws       = (float*)d_ws;
    float* attn_exp = ws;                                  // B*L*H
    float* denom    = attn_exp + (size_t)Mrows * Hh;       // B*H
    float* vbuf     = denom + Bb * Hh;                     // M*512
    float* gpool    = vbuf + (size_t)Mrows * DIMc;         // B*512
    float* posb     = gpool + Bb * DIMc;                   // B*256
    // f16 workspace (offsets are 32B-aligned)
    _Float16* hbase = (_Float16*)(posb + Bb * 256);
    _Float16* x_h   = hbase;                               // M*512
    _Float16* wv_h  = x_h + (size_t)Mrows * DIMc;          // 512*512
    _Float16* wo_h  = wv_h + DIMc * DIMc;                  // 512*512
    _Float16* opre_h = wo_h + DIMc * DIMc;                 // M*512

    // one-time fp32 -> f16 conversions
    k_cvt_f16<<<1024, 256, 0, stream>>>(x, x_h, Mrows * DIMc);
    k_cvt_f16<<<256, 256, 0, stream>>>(w_v, wv_h, DIMc * DIMc);
    k_cvt_f16<<<256, 256, 0, stream>>>(w_out, wo_h, DIMc * DIMc);

    k_attn_exp<<<Mrows, 256, 0, stream>>>(x, mask, w_attn, b_attn, attn_exp);
    k_denom<<<Bb * Hh, 256, 0, stream>>>(attn_exp, denom);
    gemm_ln_elu<<<Mrows / 16, 256, 0, stream>>>(x_h, wv_h, b_v, g_v, bb_v,
                                                nullptr, vbuf);
    k_gpool<<<Bb * Hh, 256, 0, stream>>>(attn_exp, denom, vbuf, gpool);
    k_pos<<<Bb, 256, 0, stream>>>(gpool, w_pos1, b_pos1, g_pos, bb_pos,
                                  w_pos2, b_pos2, posb);
    k_window<<<Mrows, 256, 0, stream>>>(attn_exp, posb, vbuf, gpool, opre_h);
    gemm_ln_elu<<<Mrows / 16, 256, 0, stream>>>(opre_h, wo_h, b_out, g_out,
                                                bb_out, x, out);
}
